// SSMNet_87402584473834
// MI455X (gfx1250) — matrix operand order repaired and checked by reading, not compiled, not run on hardware
//
#include <hip/hip_runtime.h>
#include <math.h>

// ---------------- problem constants (match reference) ----------------
#define T_LEN 1000
#define LEN1  500
#define LEN2  125
#define D_IN  128
#define CH1   64
#define CH2   16
#define NST   64
#define NB    128          // batch
#define STEPF 1.0e-4f      // 0.1 / 1000
#define CHK1  32           // time chunks allocated for u1 (1024/32)
#define CHK2  16           // time chunks allocated for u2 (512/32)

typedef __attribute__((ext_vector_type(16))) __bf16 v16bf;
typedef __attribute__((ext_vector_type(8)))  float  v8f;

// CDNA5 async global->LDS path (gated; plain b128 copy fallback)
#if defined(__AMDGCN__) && __has_builtin(__builtin_amdgcn_global_load_async_to_lds_b128)
#define USE_ASYNC_LDS 1
#endif

// ---------------- small helpers ----------------
__device__ __forceinline__ unsigned short f2bf(float f) {
  unsigned int x = __builtin_bit_cast(unsigned int, f);
  x += 0x7FFFu + ((x >> 16) & 1u);               // round-to-nearest-even
  return (unsigned short)(x >> 16);
}
__device__ __forceinline__ float bf2f(unsigned short u) {
  return __builtin_bit_cast(float, (unsigned int)u << 16);
}
__device__ __forceinline__ __bf16 bfbits(unsigned short u) {
  return __builtin_bit_cast(__bf16, u);
}
__device__ __forceinline__ unsigned long long pack4(unsigned short a, unsigned short b,
                                                    unsigned short c, unsigned short d) {
  return (unsigned long long)a | ((unsigned long long)b << 16) |
         ((unsigned long long)c << 32) | ((unsigned long long)d << 48);
}

struct cf { float r, i; };
__device__ __forceinline__ cf cadd(cf a, cf b) { return {a.r + b.r, a.i + b.i}; }
__device__ __forceinline__ cf cmul(cf a, cf b) {
  return {a.r * b.r - a.i * b.i, a.r * b.i + a.i * b.r};
}
__device__ __forceinline__ cf cdiv(cf a, cf b) {
  float id = 1.0f / (b.r * b.r + b.i * b.i);
  return {(a.r * b.r + a.i * b.i) * id, (a.i * b.r - a.r * b.i) * id};
}

// WMMA element->k map used here (contiguous k = e + 16*(lane/16), same as frag_slot B; wave32, bf16 16x16x32)
__device__ __forceinline__ int a_kmap(int e, int lane) {
  return e + ((lane & 16) ? 16 : 0);
}
// fragment-swizzled slot inside an 8KB (32 k x 128 b) bf16 tile:
// consumer wave wv=b>>4 reads lane (b&15)+(k&16), element k&15 contiguously
__device__ __forceinline__ int frag_slot(int b, int k) {
  return (((b >> 4) * 32 + (b & 15) + (k & 16)) * 16) + (k & 15);
}

// 8KB tile copy global->LDS (async-to-LDS when available)
__device__ __forceinline__ void tile_copy_8k(const unsigned short* __restrict__ src,
                                             unsigned short* dst, int tid) {
#ifdef USE_ASYNC_LDS
  typedef int vsi4 __attribute__((vector_size(16)));   // GCC-style int4 (builtin's pointee)
  typedef __attribute__((address_space(1))) void gvoid_t;
  typedef __attribute__((address_space(3))) void lvoid_t;
  typedef __attribute__((address_space(1))) vsi4 gvec_t;
  typedef __attribute__((address_space(3))) vsi4 lvec_t;
  for (int i = tid; i < 512; i += 256)
    __builtin_amdgcn_global_load_async_to_lds_b128(
        (gvec_t*)(gvoid_t*)(src + (size_t)i * 8),
        (lvec_t*)(lvoid_t*)(dst + (size_t)i * 8), 0, 0);
#if __has_builtin(__builtin_amdgcn_s_wait_asynccnt)
  __builtin_amdgcn_s_wait_asynccnt(0);
#else
  asm volatile("s_wait_asynccnt 0x0" ::: "memory");
#endif
#else
  for (int i = tid; i < 512; i += 256)
    *(uint4*)(dst + (size_t)i * 8) = *(const uint4*)(src + (size_t)i * 8);
#endif
}

// ---------------- stage 1: DPLR kernel at roots of unity (Cauchy) ----------------
__global__ void cauchy_k(const float* __restrict__ lr, const float* __restrict__ li,
                         const float* __restrict__ pr, const float* __restrict__ pi,
                         const float* __restrict__ br, const float* __restrict__ bi,
                         const float* __restrict__ cr, const float* __restrict__ ci,
                         float step, int L, int H, int N,
                         float* __restrict__ ar_re, float* __restrict__ ar_im) {
  int idx = blockIdx.x * blockDim.x + threadIdx.x;
  if (idx >= L * H) return;
  int l = idx % L, h = idx / L;

  float th = 6.28318530717958647692f * (float)l / (float)L;
  float sn, cs;
  sincosf(th, &sn, &cs);
  cf om   = {cs, -sn};                         // exp(-2*pi*i*l/L)
  cf onem = {1.0f - om.r, -om.i};
  cf onep = {1.0f + om.r,  om.i};
  cf g = cdiv(onem, onep);
  float sc = 2.0f / step;
  g.r *= sc; g.i *= sc;
  cf cc = cdiv(cf{2.0f, 0.0f}, onep);

  cf k00 = {0, 0}, k01 = {0, 0}, k10 = {0, 0}, k11 = {0, 0};
  for (int n = 0; n < N; ++n) {
    cf inv = cdiv(cf{1.0f, 0.0f}, cf{g.r - lr[n], g.i - li[n]});
    cf P  = {pr[n], pi[n]};
    cf Bv = {br[n], bi[n]};
    cf Cj = {cr[h * N + n], -ci[h * N + n]};   // conj(C)
    k00 = cadd(k00, cmul(cmul(Cj, Bv), inv));
    k01 = cadd(k01, cmul(cmul(Cj, P),  inv));
    cf Pj = {P.r, -P.i};                        // conj(P)
    k10 = cadd(k10, cmul(cmul(Pj, Bv), inv));
    float pp = P.r * P.r + P.i * P.i;           // conj(P)*P is real
    k11 = cadd(k11, cf{pp * inv.r, pp * inv.i});
  }
  cf r  = cdiv(k10, cf{1.0f + k11.r, k11.i});
  cf kr = cmul(k01, r);
  cf at = cmul(cc, cf{k00.r - kr.r, k00.i - kr.i});
  ar_re[h * L + l] = at.r;
  ar_im[h * L + l] = at.i;
}

// ---------------- stage 2: inverse DFT -> bf16 kernel K[h, t] ----------
__global__ void idft_k(const float* __restrict__ ar_re, const float* __restrict__ ar_im,
                       int L, int H, unsigned short* __restrict__ Kbf) {
  int idx = blockIdx.x * blockDim.x + threadIdx.x;
  if (idx >= H * L) return;
  int t = idx % L, h = idx / L;
  const float w = 6.28318530717958647692f / (float)L;
  const float* re = ar_re + h * L;
  const float* im = ar_im + h * L;
  float acc = 0.0f;
  for (int l = 0; l < L; ++l) {
    int r = (t * l) % L;                       // exact phase reduction
    float sn, cs;
    sincosf(w * (float)r, &sn, &cs);
    acc += re[l] * cs - im[l] * sn;            // Re(ar * e^{+i th})
  }
  Kbf[idx] = f2bf(acc / (float)L);
}

// ---------------- stage 3: fc1 -> u1 bf16, chunk-swizzled [64][32][4096] ----------
__global__ __launch_bounds__(256)
void fc1_k(const float* __restrict__ in, const float* __restrict__ W,
           const float* __restrict__ bias, unsigned short* __restrict__ uSw) {
  __shared__ __align__(32) unsigned short Wl[CH1 * D_IN];  // 16 KB, row-major
  __shared__ __align__(32) unsigned short Bl[32 * NB];     // 8 KB, frag-swizzled
  const int t = blockIdx.x;
  const int tid = threadIdx.x, lane = tid & 31, wv = tid >> 5;  // 8 waves

  for (int i = tid; i < CH1 * D_IN; i += 256) Wl[i] = f2bf(W[i]);

  const v8f vz = {0, 0, 0, 0, 0, 0, 0, 0};
  v8f acc[4] = {vz, vz, vz, vz};               // 4 m-tiles (h), n-tile = wv

  for (int k0 = 0; k0 < D_IN; k0 += 32) {
    __syncthreads();
    // input tile [32 d x 128 b]: float4 global load -> 4x bf16 packed b64 LDS store
    for (int i = tid; i < 1024; i += 256) {
      int b = i >> 3, dr0 = (i & 7) * 4;
      const float4 v = *(const float4*)&in[((size_t)b * T_LEN + t) * D_IN + k0 + dr0];
      *(unsigned long long*)&Bl[frag_slot(b, dr0)] =
          pack4(f2bf(v.x), f2bf(v.y), f2bf(v.z), f2bf(v.w));
    }
    __syncthreads();

    const v16bf bf = *(const v16bf*)__builtin_assume_aligned(&Bl[(wv * 32 + lane) * 16], 32);
#pragma unroll
    for (int mt = 0; mt < 4; ++mt) {
      v16bf af;
      int m = mt * 16 + (lane & 15);
#pragma unroll
      for (int e = 0; e < 16; ++e)
        af[e] = bfbits(Wl[m * D_IN + k0 + a_kmap(e, lane)]);
      acc[mt] = __builtin_amdgcn_wmma_f32_16x16x32_bf16(
          false, af, false, bf, (short)0, acc[mt], false, false);
    }
  }

  // store into chunk-swizzled layout: addr = (h*CHK1 + t/32)*4096 + frag_slot(b, t%32)
  const int bcol = wv * 16 + (lane & 15);
  const int slotb = frag_slot(bcol, t & 31);
  const int ch = t >> 5;
#pragma unroll
  for (int mt = 0; mt < 4; ++mt)
#pragma unroll
    for (int i = 0; i < 8; ++i) {
      int h = mt * 16 + i + ((lane & 16) ? 8 : 0);
      uSw[((size_t)(h * CHK1 + ch) << 12) + slotb] = f2bf(acc[mt][i] + bias[h]);
    }
}

// ---------------- stage 4/6: causal Toeplitz conv + skip + tanh + pool ----------
// uSw: bf16 [H][nalloc][4096] chunk-swizzled; Kbf: bf16 [H][L]
// mode 0: outSw = bf16 [Lp][2][4096] (channel-chunk swizzle for fc2)
// mode 1: outF  = f32  [B][Lp][H]
__global__ __launch_bounds__(256)
void conv_k(const unsigned short* __restrict__ uSw,
            const unsigned short* __restrict__ Kbf,
            const float* __restrict__ Dvec,
            unsigned short* __restrict__ outSw, float* __restrict__ outF,
            int L, int H, int pool, int Lp, int mode, int nalloc) {
  __shared__ __align__(32) unsigned short Krev[T_LEN + 128]; // reversed + zero pads
  __shared__ __align__(32) unsigned short Bl[32 * NB];       // frag-swizzled u tile
  const int h  = blockIdx.y;
  const int t0 = blockIdx.x * 64;              // 4 m-tiles per workgroup
  const int tid = threadIdx.x, lane = tid & 31, wv = tid >> 5;

  const unsigned short* u  = uSw + ((size_t)h * nalloc << 12);
  const unsigned short* Kc = Kbf + (size_t)h * L;
  // Krev[32 + (L-1-d)] = K[d]; zeros on both sides absorb d<0 and d>=L
  for (int i = tid; i < L + 128; i += 256) {
    int j = i - 32;
    Krev[i] = (j >= 0 && j < L) ? Kc[L - 1 - j] : (unsigned short)0;
  }

  const v8f vz = {0, 0, 0, 0, 0, 0, 0, 0};
  v8f acc[4] = {vz, vz, vz, vz};
  const int tmax = (t0 + 63 < L - 1) ? t0 + 63 : L - 1;
  const int nch = tmax / 32 + 1;               // triangular bound (uniform per block)

  for (int c = 0; c < nch; ++c) {
    const int s0 = c * 32;
    __syncthreads();
    tile_copy_8k(u + ((size_t)c << 12), Bl, tid);   // 8KB straight copy (async path)
    __syncthreads();
    if (c + 1 < nch)                           // -> global_prefetch_b8
      __builtin_prefetch(u + (((size_t)c + 1) << 12) + tid * 16, 0, 3);

    const v16bf bf = *(const v16bf*)__builtin_assume_aligned(&Bl[(wv * 32 + lane) * 16], 32);
#pragma unroll
    for (int mt = 0; mt < 4; ++mt) {
      if (s0 <= t0 + mt * 16 + 15) {           // uniform branch: EXEC stays all-ones
        v16bf af;                               // Toeplitz row from padded reversed K
        const int abase = 32 + (L - 1) - (t0 + mt * 16) - (lane & 15) + s0;
#pragma unroll
        for (int e = 0; e < 16; ++e) af[e] = bfbits(Krev[abase + a_kmap(e, lane)]);
        acc[mt] = __builtin_amdgcn_wmma_f32_16x16x32_bf16(
            false, af, false, bf, (short)0, acc[mt], false, false);
      }
    }
  }

  // epilogue: tanh(y + u*D) then average-pool along time
  const float Dh = Dvec[h];
  const int bcol = wv * 16 + (lane & 15);
  const int mbase = (lane & 16) ? 8 : 0;
  const int oslot = frag_slot(bcol, h & 31);   // mode-0 swizzled slot (h fixed)
  const int och = h >> 5;
#pragma unroll
  for (int mt = 0; mt < 4; ++mt) {
    float yv[8];
#pragma unroll
    for (int i = 0; i < 8; ++i) {
      int t = t0 + mt * 16 + mbase + i;
      float v = 0.0f;
      if (t < L) {
        float us = bf2f(u[((size_t)(t >> 5) << 12) + frag_slot(bcol, t & 31)]);
        v = tanhf(acc[mt][i] + us * Dh);
      }
      yv[i] = v;
    }
    for (int gph = 0; gph < 8 / pool; ++gph) {
      float s = 0.0f;
      for (int j = 0; j < pool; ++j) s += yv[gph * pool + j];
      s *= (1.0f / (float)pool);
      int p = (t0 + mt * 16 + mbase) / pool + gph;
      if (p < Lp) {
        if (mode == 0) outSw[((size_t)(p * 2 + och) << 12) + oslot] = f2bf(s);
        else           outF [((size_t)bcol * Lp + p) * H + h] = s;
      }
    }
  }
}

// ---------------- stage 5: fc2 -> u2 bf16, chunk-swizzled [16][16][4096] ----------
// h1Sw: bf16 [500][2][4096], channel-chunk swizzled (k-dim = channels)
__global__ __launch_bounds__(256)
void fc2_k(const unsigned short* __restrict__ h1Sw, const float* __restrict__ W2,
           const float* __restrict__ b2, unsigned short* __restrict__ u2Sw) {
  __shared__ __align__(32) unsigned short Wl[CH2 * CH1];     // 2 KB
  __shared__ __align__(32) unsigned short Bl[32 * NB];       // frag-swizzled
  const int t = blockIdx.x;
  const int tid = threadIdx.x, lane = tid & 31, wv = tid >> 5;

  for (int i = tid; i < CH2 * CH1; i += 256) Wl[i] = f2bf(W2[i]);

  const v8f vz = {0, 0, 0, 0, 0, 0, 0, 0};
  v8f acc = vz;
  for (int k0 = 0; k0 < CH1; k0 += 32) {
    __syncthreads();
    tile_copy_8k(h1Sw + ((size_t)(t * 2 + (k0 >> 5)) << 12), Bl, tid);
    __syncthreads();

    v16bf af;
    int m = lane & 15;
#pragma unroll
    for (int e = 0; e < 16; ++e)
      af[e] = bfbits(Wl[m * CH1 + k0 + a_kmap(e, lane)]);
    const v16bf bf = *(const v16bf*)__builtin_assume_aligned(&Bl[(wv * 32 + lane) * 16], 32);

    acc = __builtin_amdgcn_wmma_f32_16x16x32_bf16(
        false, af, false, bf, (short)0, acc, false, false);
  }

  const int bcol = wv * 16 + (lane & 15);
  const int slotb = frag_slot(bcol, t & 31);
#pragma unroll
  for (int i = 0; i < 8; ++i) {
    int h = i + ((lane & 16) ? 8 : 0);
    u2Sw[((size_t)(h * CHK2 + (t >> 5)) << 12) + slotb] = f2bf(acc[i] + b2[h]);
  }
}

// ---------------- stage 7: fc4 (tiny head, f32) ----------------
__global__ __launch_bounds__(128)
void fc4_k(const float* __restrict__ h2, const float* __restrict__ W4,
           const float* __restrict__ b4, float* __restrict__ out) {
  __shared__ float row[LEN2 * CH2];            // 2000 floats = 8 KB
  const int b = blockIdx.x, tid = threadIdx.x;
  for (int i = tid; i < LEN2 * CH2; i += blockDim.x) row[i] = h2[b * (LEN2 * CH2) + i];
  __syncthreads();
  if (tid < 20) {
    float a = b4[tid];
    const float* w = W4 + tid * (LEN2 * CH2);
    for (int f = 0; f < LEN2 * CH2; ++f) a += w[f] * row[f];
    out[b * 20 + tid] = a;
  }
}

// ---------------- host launcher ----------------
extern "C" void kernel_launch(void* const* d_in, const int* in_sizes, int n_in,
                              void* d_out, int out_size, void* d_ws, size_t ws_size,
                              hipStream_t stream) {
  (void)in_sizes; (void)n_in; (void)out_size; (void)ws_size;
  const float* input = (const float*)d_in[0];
  const float* W_fc  = (const float*)d_in[1];
  const float* b_fc  = (const float*)d_in[2];
  const float* W_fc2 = (const float*)d_in[3];
  const float* b_fc2 = (const float*)d_in[4];
  const float* W_fc4 = (const float*)d_in[5];
  const float* b_fc4 = (const float*)d_in[6];
  const float* L1r = (const float*)d_in[7],  *L1i = (const float*)d_in[8];
  const float* P1r = (const float*)d_in[9],  *P1i = (const float*)d_in[10];
  const float* B1r = (const float*)d_in[11], *B1i = (const float*)d_in[12];
  const float* C1r = (const float*)d_in[13], *C1i = (const float*)d_in[14];
  const float* D1  = (const float*)d_in[15];
  const float* L2r = (const float*)d_in[16], *L2i = (const float*)d_in[17];
  const float* P2r = (const float*)d_in[18], *P2i = (const float*)d_in[19];
  const float* B2r = (const float*)d_in[20], *B2i = (const float*)d_in[21];
  const float* C2r = (const float*)d_in[22], *C2i = (const float*)d_in[23];
  const float* D2  = (const float*)d_in[24];

  char* ws = (char*)d_ws;
  size_t off = 0;
  auto carve = [&](size_t bytes) -> void* {
    void* p = ws + off;
    off = (off + bytes + 255) & ~(size_t)255;
    return p;
  };
  float* ar1re = (float*)carve(CH1 * T_LEN * 4);
  float* ar1im = (float*)carve(CH1 * T_LEN * 4);
  float* ar2re = (float*)carve(CH2 * LEN1 * 4);
  float* ar2im = (float*)carve(CH2 * LEN1 * 4);
  unsigned short* K1b = (unsigned short*)carve(CH1 * T_LEN * 2);
  unsigned short* K2b = (unsigned short*)carve(CH2 * LEN1 * 2);
  unsigned short* u1s = (unsigned short*)carve((size_t)CH1 * CHK1 * 4096 * 2); // 16.8 MB
  unsigned short* h1s = (unsigned short*)carve((size_t)LEN1 * 2 * 4096 * 2);   //  8.2 MB
  unsigned short* u2s = (unsigned short*)carve((size_t)CH2 * CHK2 * 4096 * 2); //  2.1 MB
  float* h2 = (float*)carve((size_t)NB * LEN2 * CH2 * 4);                      //  1.0 MB

  cauchy_k<<<(T_LEN * CH1 + 255) / 256, 256, 0, stream>>>(
      L1r, L1i, P1r, P1i, B1r, B1i, C1r, C1i, STEPF, T_LEN, CH1, NST, ar1re, ar1im);
  cauchy_k<<<(LEN1 * CH2 + 255) / 256, 256, 0, stream>>>(
      L2r, L2i, P2r, P2i, B2r, B2i, C2r, C2i, STEPF, LEN1, CH2, NST, ar2re, ar2im);
  idft_k<<<(CH1 * T_LEN + 255) / 256, 256, 0, stream>>>(ar1re, ar1im, T_LEN, CH1, K1b);
  idft_k<<<(CH2 * LEN1 + 255) / 256, 256, 0, stream>>>(ar2re, ar2im, LEN1, CH2, K2b);

  fc1_k<<<T_LEN, 256, 0, stream>>>(input, W_fc, b_fc, u1s);
  conv_k<<<dim3((T_LEN + 63) / 64, CH1), 256, 0, stream>>>(
      u1s, K1b, D1, h1s, nullptr, T_LEN, CH1, 2, LEN1, 0, CHK1);
  fc2_k<<<LEN1, 256, 0, stream>>>(h1s, W_fc2, b_fc2, u2s);
  conv_k<<<dim3((LEN1 + 63) / 64, CH2), 256, 0, stream>>>(
      u2s, K2b, D2, nullptr, h2, LEN1, CH2, 4, LEN2, 1, CHK2);
  fc4_k<<<NB, 128, 0, stream>>>(h2, W_fc4, b_fc4, (float*)d_out);
}
